// GlobalAttention_48945447305897
// MI455X (gfx1250) — matrix operand order repaired and checked by reading, not compiled, not run on hardware
//
#include <hip/hip_runtime.h>
#include <hip/hip_bf16.h>

// Problem constants (from reference)
#define NB     4
#define DIMC   512
#define LQ     4096
#define HEADS  8
#define DHEAD  64
#define KW     7
#define INNER  512
#define LK     585          // (4096-7)/7 + 1
#define LKP    640          // LK padded to 5*128 (unguarded tiles)
#define KKV    (DIMC*KW)    // 3584
#define SCALE  0.125f       // 64^-0.5

typedef __bf16 bf16;
typedef __attribute__((ext_vector_type(2)))  __bf16 v2bf;
typedef __attribute__((ext_vector_type(4)))  __bf16 v4bf;
typedef __attribute__((ext_vector_type(16))) __bf16 v16bf;
typedef __attribute__((ext_vector_type(8)))  float  v8f;

static __device__ __forceinline__ v8f wmma_bf16(v16bf a, v16bf b, v8f c) {
  return __builtin_amdgcn_wmma_f32_16x16x32_bf16(false, a, false, b, (short)0, c,
                                                 false, false);
}

// CDNA5 async global->LDS copy (16B per lane), tracked by ASYNCcnt.
static __device__ __forceinline__ void async_copy_b128(void* lds, const void* gptr) {
  unsigned lds_off = (unsigned)(size_t)lds;   // generic LDS addr low 32 = LDS offset
  asm volatile("global_load_async_to_lds_b128 %0, %1, off"
               :: "v"(lds_off), "v"(gptr) : "memory");
}
static __device__ __forceinline__ void wait_async() {
  asm volatile("s_wait_asynccnt 0" ::: "memory");
}

// A fragment (16x32 bf16, MxK): lane l holds row m=l&15; K = two 8-elem runs:
//   lanes 0-15: K 0..7 and 16..23 ; lanes 16-31: K 8..15 and 24..31
static __device__ __forceinline__ v16bf load_a_frag(const bf16* s, int ldk, int lane) {
  int m   = lane & 15;
  int kb  = (lane < 16) ? 0 : 8;
  const bf16* p0 = s + m * ldk + kb;        // 8 contiguous (16B)
  const bf16* p1 = p0 + 16;                 // 8 contiguous (16B)
  v16bf a;
#pragma unroll
  for (int i = 0; i < 8; ++i) { a[i] = p0[i]; a[i + 8] = p1[i]; }
  return a;
}

// B fragment (32x16 bf16, KxN) from an LDS tile stored TRANSPOSED as [n][k]:
//   lane l holds col n=l&15; K = two 8-elem runs (lanes 0-15: K0..7,16..23; 16-31: K8..15,24..31)
static __device__ __forceinline__ v16bf load_b_frag(const bf16* s, int ldk, int lane) {
  int n  = lane & 15;
  int k0 = (lane < 16) ? 0 : 8;
  const bf16* p = s + n * ldk + k0;         // two 8-elem runs
  v16bf b;
#pragma unroll
  for (int i = 0; i < 8; ++i) { b[i] = p[i]; b[i + 8] = p[i + 16]; }
  return b;
}

// ---------------------------------------------------------------- conversions
__global__ __launch_bounds__(256) void f32_to_bf16_kernel(
    const float* __restrict__ in, bf16* __restrict__ out, int n) {
  int i = (blockIdx.x * 256 + threadIdx.x) * 4;
  if (i < n) {
    float4 v = *(const float4*)(in + i);
    v4bf o; o[0] = (bf16)v.x; o[1] = (bf16)v.y; o[2] = (bf16)v.z; o[3] = (bf16)v.w;
    *(v4bf*)(out + i) = o;
  }
}

// ------------------------------------------------------- generic WMMA GEMM NN
// C[bz, m, n] = sum_k A[m,k] * B[bz, k, n];  A: MxKd row-major (weights, shared),
// B batch stride Kd*N.  N must be a multiple of 128 (callers guarantee).
// WG tile 128M x 128N; 8 waves as 4(M)x2(N); wave tile 32M x 64N -> 8 WMMA/k-step.
// MODE 0: bf16 out.  MODE 1: f32 out + bias[m].
template <int MODE>
__global__ __launch_bounds__(256) void gemm_nn(
    const bf16* __restrict__ A, const bf16* __restrict__ Bm,
    bf16* __restrict__ outB, float* __restrict__ outF,
    const float* __restrict__ bias, int M, int N, int Kd) {
  __shared__ bf16 sA[128][32];      // [m][k]
  __shared__ bf16 sB[128][40];      // [n][k] transposed, padded
  const int tid = threadIdx.x, wave = tid >> 5, lane = tid & 31;
  const int wm = wave >> 1, wn = wave & 1;
  const int m0 = blockIdx.y * 128, n0 = blockIdx.x * 128, bz = blockIdx.z;
  const bf16* Bb = Bm + (size_t)bz * Kd * N;
  v8f acc[2][4] = {};

  const int ar = tid >> 1, ac = (tid & 1) * 16;        // A-tile assignment
  const int k2 = (tid >> 4) << 1, nb = (tid & 15) * 8; // B-tile assignment

  for (int k0 = 0; k0 < Kd; k0 += 32) {
    // A tile 128x32: async global->LDS, 2x16B per thread
    {
      const bf16* src = A + (size_t)(m0 + ar) * Kd + k0 + ac;
      async_copy_b128(&sA[ar][ac], src);
      async_copy_b128(&sA[ar][ac + 8], src + 8);
    }
    // B tile 32x128 -> transposed sB[n][k]: 2 k-rows x 8 n per thread, b32 stores
    {
      const bf16* s0 = Bb + (size_t)(k0 + k2) * N + n0 + nb;
      const bf16* s1 = s0 + N;
#pragma unroll
      for (int i = 0; i < 8; ++i) {
        v2bf p; p[0] = s0[i]; p[1] = s1[i];
        *(v2bf*)&sB[nb + i][k2] = p;
      }
    }
    wait_async();
    __syncthreads();
    v16bf a0 = load_a_frag(&sA[wm * 32][0], 32, lane);
    v16bf a1 = load_a_frag(&sA[wm * 32 + 16][0], 32, lane);
#pragma unroll
    for (int nt = 0; nt < 4; ++nt) {
      v16bf b = load_b_frag(&sB[wn * 64 + nt * 16][0], 40, lane);
      acc[0][nt] = wmma_bf16(a0, b, acc[0][nt]);
      acc[1][nt] = wmma_bf16(a1, b, acc[1][nt]);
    }
    __syncthreads();
  }
  // C layout: lane holds col = lane&15; rows r + (lane<16?0:8), r = 0..7
  const int col = lane & 15, rbase = (lane < 16) ? 0 : 8;
#pragma unroll
  for (int h = 0; h < 2; ++h) {
#pragma unroll
    for (int nt = 0; nt < 4; ++nt) {
      int on = n0 + wn * 64 + nt * 16 + col;
#pragma unroll
      for (int r = 0; r < 8; ++r) {
        int om = m0 + wm * 32 + h * 16 + rbase + r;
        size_t idx = (size_t)bz * M * N + (size_t)om * N + on;
        if (MODE == 0) outB[idx] = (bf16)acc[h][nt][r];
        else           outF[idx] = acc[h][nt][r] + bias[om];
      }
    }
  }
}

// ----------------------------------------------- stride-7 conv (kv) as GEMM
// kv[bz, o, j] = sum_{u=c*7+t} Wkv[o,u] * X[bz, c, j*7+t]
// M=1024, N=LKP(640, padded), K=3584.  X has a padded tail so the im2col
// gather needs no bounds checks; padded output columns hold garbage that is
// exactly masked out downstream.
__global__ __launch_bounds__(256) void gemm_kv(
    const bf16* __restrict__ A, const bf16* __restrict__ X,
    bf16* __restrict__ out) {
  __shared__ bf16 sA[128][32];
  __shared__ bf16 sB[128][40];
  const int tid = threadIdx.x, wave = tid >> 5, lane = tid & 31;
  const int wm = wave >> 1, wn = wave & 1;
  const int m0 = blockIdx.y * 128, n0 = blockIdx.x * 128, bz = blockIdx.z;
  const bf16* Xb = X + (size_t)bz * DIMC * LQ;
  v8f acc[2][4] = {};

  const int ar = tid >> 1, ac = (tid & 1) * 16;
  const int k2 = (tid >> 4) << 1, nb = (tid & 15) * 8;

  for (int k0 = 0; k0 < KKV; k0 += 32) {
    {   // A tile via async copy
      const bf16* src = A + (size_t)(m0 + ar) * KKV + k0 + ac;
      async_copy_b128(&sA[ar][ac], src);
      async_copy_b128(&sA[ar][ac + 8], src + 8);
    }
    {   // im2col gather into transposed B tile (unguarded; X tail-padded)
      int u0 = k0 + k2, u1 = u0 + 1;
      int c0 = u0 / KW, t0 = u0 % KW;
      int c1 = u1 / KW, t1 = u1 % KW;
      const bf16* g0 = Xb + (size_t)c0 * LQ + t0 + (size_t)(n0 + nb) * KW;
      const bf16* g1 = Xb + (size_t)c1 * LQ + t1 + (size_t)(n0 + nb) * KW;
#pragma unroll
      for (int i = 0; i < 8; ++i) {
        v2bf p; p[0] = g0[(size_t)i * KW]; p[1] = g1[(size_t)i * KW];
        *(v2bf*)&sB[nb + i][k2] = p;
      }
    }
    wait_async();
    __syncthreads();
    v16bf a0 = load_a_frag(&sA[wm * 32][0], 32, lane);
    v16bf a1 = load_a_frag(&sA[wm * 32 + 16][0], 32, lane);
#pragma unroll
    for (int nt = 0; nt < 4; ++nt) {
      v16bf b = load_b_frag(&sB[wn * 64 + nt * 16][0], 40, lane);
      acc[0][nt] = wmma_bf16(a0, b, acc[0][nt]);
      acc[1][nt] = wmma_bf16(a1, b, acc[1][nt]);
    }
    __syncthreads();
  }
  const int col = lane & 15, rbase = (lane < 16) ? 0 : 8;
#pragma unroll
  for (int h = 0; h < 2; ++h) {
#pragma unroll
    for (int nt = 0; nt < 4; ++nt) {
      int on = n0 + wn * 64 + nt * 16 + col;
#pragma unroll
      for (int r = 0; r < 8; ++r) {
        int om = m0 + wm * 32 + h * 16 + rbase + r;
        out[(size_t)bz * 2 * INNER * LKP + (size_t)om * LKP + on] =
            (bf16)acc[h][nt][r];
      }
    }
  }
}

// --------------------------------------------------------- flash attention
// q:(B,512,4096) bf16 [d-major rows, l cols]; kv:(B,1024,LKP); att:(B,512,4096)
__global__ __launch_bounds__(256) void attention_kernel(
    const bf16* __restrict__ q, const bf16* __restrict__ kv,
    bf16* __restrict__ att) {
  __shared__ bf16 sK[32][64];        // [j_local][d]  (transposed K chunk)
  __shared__ bf16 sV[64][32];        // [d][j_local]  (straight V chunk)
  __shared__ bf16 sP[8][16][32];     // per-wave P staging (C-layout -> A-layout)
  const int tid = threadIdx.x, wave = tid >> 5, lane = tid & 31;
  const int b = blockIdx.y >> 3, h = blockIdx.y & 7;
  const int l0 = blockIdx.x * 128 + wave * 16;
  const int mcol = lane & 15, hi8 = (lane < 16) ? 0 : 8;

  // Q A-fragments (16 rows x 64 d = 2 k-chunks), loaded once, scattered global
  const bf16* qbase = q + ((size_t)b * INNER + h * DHEAD) * LQ;
  v16bf qa[2];
#pragma unroll
  for (int kc = 0; kc < 2; ++kc) {
#pragma unroll
    for (int i = 0; i < 16; ++i) {
      int d = kc * 32 + (i >> 3) * 16 + hi8 + (i & 7);
      qa[kc][i] = qbase[(size_t)d * LQ + l0 + mcol];
    }
  }

  v8f oacc[4] = {};
  float mrow[8], lrow[8];
#pragma unroll
  for (int r = 0; r < 8; ++r) { mrow[r] = -1e30f; lrow[r] = 0.0f; }

  const bf16* Kb = kv + ((size_t)b * 2 * INNER + h * DHEAD) * LKP;
  const bf16* Vb = kv + ((size_t)b * 2 * INNER + INNER + h * DHEAD) * LKP;
  const int kj = tid >> 3, kd = (tid & 7) * 8;   // K-chunk assignment
  const int vd = tid >> 2, vj = (tid & 3) * 8;   // V-chunk assignment

  for (int j0 = 0; j0 < 608; j0 += 32) {      // ceil(585/32)=19 chunks
    // V chunk: straight copy -> async global->LDS (16B per thread)
    async_copy_b128(&sV[vd][vj], Vb + (size_t)vd * LKP + j0 + vj);
    // K chunk: transpose-gather into sK[j][d] (unguarded: kv is padded)
    {
      const bf16* src = Kb + (size_t)kd * LKP + j0 + kj;
#pragma unroll
      for (int i = 0; i < 8; ++i) sK[kj][kd + i] = src[(size_t)i * LKP];
    }
    wait_async();
    __syncthreads();

    // S = Q K^T   (two 16x16 n-tiles, contraction d=64 in two k-chunks)
    v8f s0 = {}, s1 = {};
#pragma unroll
    for (int kc = 0; kc < 2; ++kc) {
      v16bf b0 = load_b_frag(&sK[0][kc * 32], 64, lane);
      s0 = wmma_bf16(qa[kc], b0, s0);
      v16bf b1 = load_b_frag(&sK[16][kc * 32], 64, lane);
      s1 = wmma_bf16(qa[kc], b1, s1);
    }

    // online softmax (row stats per C-fragment row r; cols across 16 lanes)
    const int jc0 = j0 + mcol, jc1 = j0 + 16 + mcol;
#pragma unroll
    for (int r = 0; r < 8; ++r) {
      float a0 = (jc0 < LK) ? s0[r] * SCALE : -1e30f;   // mask padded logits
      float a1 = (jc1 < LK) ? s1[r] * SCALE : -1e30f;
      float vmax = fmaxf(a0, a1);
#pragma unroll
      for (int off = 8; off >= 1; off >>= 1)
        vmax = fmaxf(vmax, __shfl_xor(vmax, off, 32));
      float nm    = fmaxf(mrow[r], vmax);
      float alpha = __expf(mrow[r] - nm);
      float p0 = __expf(a0 - nm), p1 = __expf(a1 - nm); // ==0 for masked cols
      float sum = p0 + p1;
#pragma unroll
      for (int off = 8; off >= 1; off >>= 1)
        sum += __shfl_xor(sum, off, 32);
      lrow[r] = lrow[r] * alpha + sum;
      mrow[r] = nm;
#pragma unroll
      for (int dt = 0; dt < 4; ++dt) oacc[dt][r] *= alpha;
      int prow = r + hi8;
      sP[wave][prow][mcol]      = (bf16)p0;
      sP[wave][prow][16 + mcol] = (bf16)p1;
    }
    asm volatile("s_wait_dscnt 0" ::: "memory");   // in-wave LDS ordering for sP

    // O += P V   (contraction over 32 j; 4 d-tiles; garbage V cols hit P==0)
    v16bf pa = load_a_frag(&sP[wave][0][0], 32, lane);
#pragma unroll
    for (int dt = 0; dt < 4; ++dt) {
      v16bf bv = load_b_frag(&sV[dt * 16][0], 32, lane);
      oacc[dt] = wmma_bf16(pa, bv, oacc[dt]);
    }
    __syncthreads();
  }

  // normalize (v_rcp + mul) and store (att layout: (b, h*64+d, l))
  float rinv[8];
#pragma unroll
  for (int r = 0; r < 8; ++r) rinv[r] = __builtin_amdgcn_rcpf(lrow[r]);
  bf16* ob = att + ((size_t)b * INNER + h * DHEAD) * LQ;
#pragma unroll
  for (int dt = 0; dt < 4; ++dt) {
#pragma unroll
    for (int r = 0; r < 8; ++r) {
      int d = dt * 16 + mcol, l = l0 + r + hi8;
      ob[(size_t)d * LQ + l] = (bf16)(oacc[dt][r] * rinv[r]);
    }
  }
}

// --------------------------------------------------------------------- host
extern "C" void kernel_launch(void* const* d_in, const int* in_sizes, int n_in,
                              void* d_out, int out_size, void* d_ws, size_t ws_size,
                              hipStream_t stream) {
  (void)in_sizes; (void)n_in; (void)out_size; (void)ws_size;
  const float* x   = (const float*)d_in[0];
  const float* wq  = (const float*)d_in[1];
  const float* wkv = (const float*)d_in[2];
  const float* wo  = (const float*)d_in[3];
  const float* bo  = (const float*)d_in[4];
  float* out = (float*)d_out;

  char* ws = (char*)d_ws;
  size_t off = 0;
  auto take = [&](size_t elems) {
    void* p = ws + off;
    off = (off + elems * sizeof(bf16) + 255) & ~(size_t)255;
    return (bf16*)p;
  };
  const size_t nX   = (size_t)NB * DIMC * LQ;       // 8.4M
  const size_t nWq  = (size_t)INNER * DIMC;
  const size_t nWkv = (size_t)2 * INNER * DIMC * KW;
  const size_t nWo  = (size_t)DIMC * INNER;
  const size_t nKV  = (size_t)NB * 2 * INNER * LKP; // padded Lk

  bf16* xh   = take(nX + 1024);   // tail pad: unguarded im2col overshoot
  bf16* wqh  = take(nWq);
  bf16* wkvh = take(nWkv);
  bf16* woh  = take(nWo);
  bf16* qb   = take(nX);
  bf16* kvb  = take(nKV);
  bf16* attb = take(nX);

  auto cvt = [&](const float* src, bf16* dst, size_t n) {
    unsigned blocks = (unsigned)((n / 4 + 255) / 256);
    f32_to_bf16_kernel<<<dim3(blocks), dim3(256), 0, stream>>>(src, dst, (int)n);
  };
  cvt(x, xh, nX);  cvt(wq, wqh, nWq);  cvt(wkv, wkvh, nWkv);  cvt(wo, woh, nWo);

  // q = wq * x    (bf16 out)
  gemm_nn<0><<<dim3(LQ / 128, INNER / 128, NB), dim3(256), 0, stream>>>(
      wqh, xh, qb, nullptr, nullptr, INNER, LQ, DIMC);
  // kv = wkv *_s7 x   (padded N = LKP)
  gemm_kv<<<dim3(LKP / 128, (2 * INNER) / 128, NB), dim3(256), 0, stream>>>(
      wkvh, xh, kvb);
  // attention
  attention_kernel<<<dim3(LQ / 128, NB * HEADS), dim3(256), 0, stream>>>(
      qb, kvb, attb);
  // out = wo * att + bo   (f32 out)
  gemm_nn<1><<<dim3(LQ / 128, DIMC / 128, NB), dim3(256), 0, stream>>>(
      woh, attb, nullptr, out, bo, DIMC, LQ, INNER);
}
